// DeepSeekMoeWrapperCached_44418551775992
// MI455X (gfx1250) — matrix operand order repaired and checked by reading, not compiled, not run on hardware
//
#include <hip/hip_runtime.h>
#include <hip/hip_bf16.h>

// DeepSeek MoE for MI455X (gfx1250, wave32, WMMA).
// bf16 V_WMMA_F32_16X16X32_BF16 GEMMs, f32 accumulate. Routed experts as a
// gather/grouped GEMM over exactly T*K = 8192 token-slots (4x fewer FLOPs
// than dense expert scan). 64x128 tiles: each wave owns 16x64 of C.
// B fragments are preloaded in batches of 4 so each K-step is
// (8x ds_load_b128, one s_wait, 4x WMMA) x2 instead of a wait per WMMA.

#define T_TOK 2048
#define D_HID 1024
#define N_EXP 16
#define K_TOP 4
#define F_INT 1024
#define SHF   2048
#define R_SCALE 2.5f

typedef __attribute__((ext_vector_type(16))) __bf16 v16bf;
typedef __attribute__((ext_vector_type(8)))  __bf16 v8bf;
typedef __attribute__((ext_vector_type(8)))  float  v8f;

#define LDT 40  // LDS row stride in bf16 elements (80 B, 16 B aligned)

// A fragment, 16x32 bf16, CDNA5 ISA layout:
// lanes 0-15:  M=lane,    VGPR0-3 K=0..7,  VGPR4-7 K=16..23
// lanes 16-31: M=lane-16, VGPR0-3 K=8..15, VGPR4-7 K=24..31
__device__ __forceinline__ v16bf frag_a(const __bf16* lds, int m, int lane) {
  const int kh = lane >> 4;
  v8bf lo = *(const v8bf*)&lds[m * LDT + kh * 8];
  v8bf hi = *(const v8bf*)&lds[m * LDT + 16 + kh * 8];
  v16bf r;
#pragma unroll
  for (int i = 0; i < 8; ++i) { r[i] = lo[i]; r[i + 8] = hi[i]; }
  return r;
}

// B fragment, 32x16 bf16: lanes 0-15 hold K=0..7,16..23 at N=lane&15,
// lanes 16-31 hold K=8..15,24..31. LDS holds B transposed [N][K] so each lane
// reads two runs of 8 contiguous bf16 (two ds_load_b128).
__device__ __forceinline__ v16bf frag_b(const __bf16* ldsT, int n, int lane) {
  const int kh = lane >> 4;
  v8bf lo = *(const v8bf*)&ldsT[n * LDT + kh * 8];
  v8bf hi = *(const v8bf*)&ldsT[n * LDT + 16 + kh * 8];
  v16bf r;
#pragma unroll
  for (int i = 0; i < 8; ++i) { r[i] = lo[i]; r[i + 8] = hi[i]; }
  return r;
}

// Stage one 32(K) x 128(N) fp32 weight slab into LDS transposed [n][k] bf16.
// Thread owns column n = tid&127, K-half (tid>>7)*16: 16 coalesced
// global_load_b32 (consecutive tid -> consecutive n) + two ds_store_b128.
__device__ __forceinline__ void stage_b(const float* __restrict__ W, int Fn,
                                        int k0, int n0, __bf16* ldsT, int tid) {
  const int bn = tid & 127;
  const int bk = (tid >> 7) * 16;
  const float* src = &W[(size_t)(k0 + bk) * Fn + n0 + bn];
  v8bf lo, hi;
#pragma unroll
  for (int i = 0; i < 8; ++i) {
    lo[i] = (__bf16)src[(size_t)i * Fn];
    hi[i] = (__bf16)src[(size_t)(i + 8) * Fn];
  }
  *(v8bf*)&ldsT[bn * LDT + bk]     = lo;
  *(v8bf*)&ldsT[bn * LDT + bk + 8] = hi;
}

__device__ __forceinline__ float silu_f(float x) { return x / (1.f + __expf(-x)); }

// ---------------------------------------------------------------- convert
__global__ __launch_bounds__(256) void k_convert(const float* __restrict__ x,
                                                 __bf16* __restrict__ xb) {
  int i = (blockIdx.x * 256 + threadIdx.x) * 4;
  float4 v = *(const float4*)&x[i];
  xb[i + 0] = (__bf16)v.x; xb[i + 1] = (__bf16)v.y;
  xb[i + 2] = (__bf16)v.z; xb[i + 3] = (__bf16)v.w;
}

// ---------------------------------------------------------------- gate/topk
__global__ __launch_bounds__(256) void k_gate(const float* __restrict__ x,
                                              const float* __restrict__ gw,
                                              int* __restrict__ topk_idx,
                                              float* __restrict__ topk_w) {
  const int lane = threadIdx.x & 31;
  const int wave = threadIdx.x >> 5;
  const int t = blockIdx.x * 8 + wave;        // one wave32 per token
  const int e = lane & 15, half = lane >> 4;  // 2 lanes per expert
  const float* xr = x  + (size_t)t * D_HID + half * 512;
  const float* wr = gw + (size_t)e * D_HID + half * 512;
  float s = 0.f;
  for (int d = 0; d < 512; ++d) s += xr[d] * wr[d];
  s += __shfl_xor(s, 16, 32);
  float score = 1.f / (1.f + __expf(-s));     // sigmoid
  float sc[N_EXP];
#pragma unroll
  for (int i = 0; i < N_EXP; ++i) sc[i] = __shfl(score, i, 32);
  int idx[K_TOP]; float w[K_TOP]; float sum = 0.f;
#pragma unroll
  for (int j = 0; j < K_TOP; ++j) {
    int bi = 0; float bv = -1.f;
#pragma unroll
    for (int i = 0; i < N_EXP; ++i) if (sc[i] > bv) { bv = sc[i]; bi = i; }
    idx[j] = bi; w[j] = bv; sum += bv; sc[bi] = -1.f;
  }
  if (lane == 0) {
    float inv = R_SCALE / (sum + 1e-20f);
#pragma unroll
    for (int j = 0; j < K_TOP; ++j) {
      topk_idx[t * K_TOP + j] = idx[j];
      topk_w[t * K_TOP + j]   = w[j] * inv;
    }
  }
}

// ---------------------------------------------------------------- bucket count
__global__ __launch_bounds__(256) void k_count(const int* __restrict__ topk_idx,
                                               int* __restrict__ counts,
                                               int* __restrict__ offsets,
                                               int* __restrict__ cursor) {
  int tid = threadIdx.x;
  if (tid < N_EXP) counts[tid] = 0;
  __syncthreads();
  for (int s = tid; s < T_TOK * K_TOP; s += 256)
    atomicAdd(&counts[topk_idx[s]], 1);
  __syncthreads();
  if (tid == 0) {
    int acc = 0;
    for (int e = 0; e < N_EXP; ++e) { offsets[e] = acc; cursor[e] = acc; acc += counts[e]; }
    offsets[N_EXP] = acc;  // == T_TOK*K_TOP
  }
}

// ---------------------------------------------------------------- scatter slots
__global__ __launch_bounds__(256) void k_scatter(const int* __restrict__ topk_idx,
                                                 const float* __restrict__ topk_w,
                                                 int* __restrict__ cursor,
                                                 int* __restrict__ slot_tok,
                                                 float* __restrict__ slot_w) {
  int s = blockIdx.x * 256 + threadIdx.x;
  int e = topk_idx[s];
  int p = atomicAdd(&cursor[e], 1);
  slot_tok[p] = s >> 2;         // token id
  slot_w[p]   = topk_w[s];
}

// ---------------------------------------------------------------- gate+up GEMM
// H[row, n] = silu(A @ Wg)[row, n] * (A @ Wu)[row, n], bf16 output.
// Block: 256 thr = 8 waves (4 M x 2 N), tile 64x128, K step 32.
template <bool ROUTED>
__global__ __launch_bounds__(256) void k_gateup(
    const __bf16* __restrict__ xb, const float* __restrict__ Wg_all,
    const float* __restrict__ Wu_all, __bf16* __restrict__ H,
    const int* __restrict__ offsets, const int* __restrict__ slot_tok,
    int Kd, int Fn) {
  const int e = ROUTED ? blockIdx.z : 0;
  int base = 0, rows = T_TOK;
  if (ROUTED) { base = offsets[e]; rows = offsets[e + 1] - base; }
  const int m0 = blockIdx.y * 64;
  if (m0 >= rows) return;  // uniform exit: EXEC stays all-ones for WMMA
  const int n0 = blockIdx.x * 128;
  const float* Wg = Wg_all + (size_t)e * Kd * Fn;
  const float* Wu = Wu_all + (size_t)e * Kd * Fn;

  __shared__ __bf16 As[64 * LDT];    //  5.1 KB
  __shared__ __bf16 BgT[128 * LDT];  // 10.2 KB
  __shared__ __bf16 BuT[128 * LDT];  // 10.2 KB

  const int tid = threadIdx.x, lane = tid & 31, wave = tid >> 5;
  const int wm = wave >> 1, wn = wave & 1;   // wave tile: rows wm*16, cols wn*64

  v8f accG[4], accU[4];
#pragma unroll
  for (int j = 0; j < 4; ++j)
#pragma unroll
    for (int i = 0; i < 8; ++i) { accG[j][i] = 0.f; accU[j][i] = 0.f; }

  const int mrow = tid >> 2, ko = tid & 3;   // A stage: 64 rows x 4 octets
  const bool valid = (m0 + mrow) < rows;
  long arow = m0 + mrow;
  if (ROUTED) arow = valid ? (long)slot_tok[base + m0 + mrow] : 0;

  for (int k0 = 0; k0 < Kd; k0 += 32) {
    v8bf av;
#pragma unroll
    for (int i = 0; i < 8; ++i) av[i] = (__bf16)0.f;
    if (valid) av = *(const v8bf*)&xb[arow * Kd + k0 + ko * 8];
    *(v8bf*)&As[mrow * LDT + ko * 8] = av;

    stage_b(Wg, Fn, k0, n0, BgT, tid);
    stage_b(Wu, Fn, k0, n0, BuT, tid);
    if (k0 + 32 < Kd) {                       // global_prefetch next K slab
      __builtin_prefetch(&Wg[(size_t)(k0 + 32 + (tid >> 7) * 16) * Fn + n0 + (tid & 127)], 0, 0);
      __builtin_prefetch(&Wu[(size_t)(k0 + 32 + (tid >> 7) * 16) * Fn + n0 + (tid & 127)], 0, 0);
    }
    __syncthreads();

    v16bf a = frag_a(As, wm * 16 + (lane & 15), lane);

    // Batch-load 4 G fragments, one wait, then 4 back-to-back WMMAs.
    v16bf bg[4];
#pragma unroll
    for (int j = 0; j < 4; ++j)
      bg[j] = frag_b(BgT, wn * 64 + j * 16 + (lane & 15), lane);
#pragma unroll
    for (int j = 0; j < 4; ++j)
      accG[j] = __builtin_amdgcn_wmma_f32_16x16x32_bf16(false, a, false, bg[j],
                                                        (short)0, accG[j], false, false);
    // Same for U fragments.
    v16bf bu[4];
#pragma unroll
    for (int j = 0; j < 4; ++j)
      bu[j] = frag_b(BuT, wn * 64 + j * 16 + (lane & 15), lane);
#pragma unroll
    for (int j = 0; j < 4; ++j)
      accU[j] = __builtin_amdgcn_wmma_f32_16x16x32_bf16(false, a, false, bu[j],
                                                        (short)0, accU[j], false, false);
    __syncthreads();
  }

  // C layout: VGPR r, lanes 0-15 -> M=r, lanes 16-31 -> M=r+8; N=lane&15
#pragma unroll
  for (int j = 0; j < 4; ++j) {
#pragma unroll
    for (int r = 0; r < 8; ++r) {
      int m = wm * 16 + r + ((lane >> 4) << 3);
      if (m0 + m < rows) {
        int n = n0 + wn * 64 + j * 16 + (lane & 15);
        float h = silu_f(accG[j][r]) * accU[j][r];
        H[(size_t)(base + m0 + m) * Fn + n] = (__bf16)h;
      }
    }
  }
}

// ---------------------------------------------------------------- down GEMM
template <bool ROUTED>
__global__ __launch_bounds__(256) void k_down(
    const __bf16* __restrict__ Hbuf, const float* __restrict__ Wd_all,
    float* __restrict__ out, const int* __restrict__ offsets,
    const int* __restrict__ slot_tok, const float* __restrict__ slot_w,
    int Kd, int Dn) {
  const int e = ROUTED ? blockIdx.z : 0;
  int base = 0, rows = T_TOK;
  if (ROUTED) { base = offsets[e]; rows = offsets[e + 1] - base; }
  const int m0 = blockIdx.y * 64;
  if (m0 >= rows) return;
  const int n0 = blockIdx.x * 128;
  const float* Wd = Wd_all + (size_t)e * Kd * Dn;

  __shared__ __bf16 As[64 * LDT];
  __shared__ __bf16 BdT[128 * LDT];

  const int tid = threadIdx.x, lane = tid & 31, wave = tid >> 5;
  const int wm = wave >> 1, wn = wave & 1;

  v8f acc[4];
#pragma unroll
  for (int j = 0; j < 4; ++j)
#pragma unroll
    for (int i = 0; i < 8; ++i) acc[j][i] = 0.f;

  const int mrow = tid >> 2, ko = tid & 3;
  const bool valid = (m0 + mrow) < rows;
  const size_t arow = (size_t)(base + (valid ? m0 + mrow : 0));

  for (int k0 = 0; k0 < Kd; k0 += 32) {
    v8bf av;
#pragma unroll
    for (int i = 0; i < 8; ++i) av[i] = (__bf16)0.f;
    if (valid) av = *(const v8bf*)&Hbuf[arow * Kd + k0 + ko * 8];
    *(v8bf*)&As[mrow * LDT + ko * 8] = av;

    stage_b(Wd, Dn, k0, n0, BdT, tid);
    if (k0 + 32 < Kd)
      __builtin_prefetch(&Wd[(size_t)(k0 + 32 + (tid >> 7) * 16) * Dn + n0 + (tid & 127)], 0, 0);
    __syncthreads();

    v16bf a = frag_a(As, wm * 16 + (lane & 15), lane);
    v16bf b[4];
#pragma unroll
    for (int j = 0; j < 4; ++j)
      b[j] = frag_b(BdT, wn * 64 + j * 16 + (lane & 15), lane);
#pragma unroll
    for (int j = 0; j < 4; ++j)
      acc[j] = __builtin_amdgcn_wmma_f32_16x16x32_bf16(false, a, false, b[j],
                                                       (short)0, acc[j], false, false);
    __syncthreads();
  }

#pragma unroll
  for (int j = 0; j < 4; ++j) {
#pragma unroll
    for (int r = 0; r < 8; ++r) {
      int m = wm * 16 + r + ((lane >> 4) << 3);
      if (m0 + m < rows) {
        int n = n0 + wn * 64 + j * 16 + (lane & 15);
        if (ROUTED) {
          int t = slot_tok[base + m0 + m];
          float w = slot_w[base + m0 + m];
          atomicAdd(&out[(size_t)t * Dn + n], acc[j][r] * w);  // global_atomic_add_f32
        } else {
          out[(size_t)(m0 + m) * Dn + n] = acc[j][r];          // shared path writes first
        }
      }
    }
  }
}

// ---------------------------------------------------------------- launch
extern "C" void kernel_launch(void* const* d_in, const int* in_sizes, int n_in,
                              void* d_out, int out_size, void* d_ws, size_t ws_size,
                              hipStream_t stream) {
  (void)in_sizes; (void)n_in; (void)out_size; (void)ws_size;
  const float* x   = (const float*)d_in[0];
  const float* gw  = (const float*)d_in[1];
  const float* swg = (const float*)d_in[2];
  const float* swu = (const float*)d_in[3];
  const float* swd = (const float*)d_in[4];
  const float* ewg = (const float*)d_in[5];
  const float* ewu = (const float*)d_in[6];
  const float* ewd = (const float*)d_in[7];
  float* out = (float*)d_out;

  char* ws = (char*)d_ws;
  size_t o = 0;
  __bf16* xb = (__bf16*)(ws + o); o += (size_t)T_TOK * D_HID * 2;          // 4 MB
  __bf16* hs = (__bf16*)(ws + o); o += (size_t)T_TOK * SHF * 2;            // 8 MB
  __bf16* hr = (__bf16*)(ws + o); o += (size_t)T_TOK * K_TOP * F_INT * 2;  // 16 MB
  int*   topk_idx = (int*)(ws + o);   o += (size_t)T_TOK * K_TOP * 4;
  float* topk_w   = (float*)(ws + o); o += (size_t)T_TOK * K_TOP * 4;
  int*   counts   = (int*)(ws + o);   o += 256;
  int*   offsets  = (int*)(ws + o);   o += 256;
  int*   cursor   = (int*)(ws + o);   o += 256;
  int*   slot_tok = (int*)(ws + o);   o += (size_t)T_TOK * K_TOP * 4;
  float* slot_w   = (float*)(ws + o); o += (size_t)T_TOK * K_TOP * 4;

  k_convert<<<(T_TOK * D_HID) / 1024, 256, 0, stream>>>(x, xb);
  k_gate<<<T_TOK / 8, 256, 0, stream>>>(x, gw, topk_idx, topk_w);
  k_count<<<1, 256, 0, stream>>>(topk_idx, counts, offsets, cursor);
  k_scatter<<<(T_TOK * K_TOP) / 256, 256, 0, stream>>>(topk_idx, topk_w, cursor,
                                                       slot_tok, slot_w);
  // shared expert (writes out), then routed experts (atomic accumulate)
  k_gateup<false><<<dim3(SHF / 128, T_TOK / 64, 1), 256, 0, stream>>>(
      xb, swg, swu, hs, nullptr, nullptr, D_HID, SHF);
  k_down<false><<<dim3(D_HID / 128, T_TOK / 64, 1), 256, 0, stream>>>(
      hs, swd, out, nullptr, nullptr, nullptr, SHF, D_HID);
  k_gateup<true><<<dim3(F_INT / 128, T_TOK / 64, N_EXP), 256, 0, stream>>>(
      xb, ewg, ewu, hr, offsets, slot_tok, D_HID, F_INT);
  k_down<true><<<dim3(D_HID / 128, T_TOK / 64, N_EXP), 256, 0, stream>>>(
      hr, ewd, out, offsets, slot_tok, slot_w, F_INT, D_HID);
}
